// Prototype_Layer_74199855005949
// MI455X (gfx1250) — matrix operand order repaired and checked by reading, not compiled, not run on hardware
//
#include <hip/hip_runtime.h>
#include <stdint.h>

// ---------------------------------------------------------------------------
// Prototype layer: out[b,p,h,w] = <xn[b,:,h,w], pn[p,:]>  (cosine similarity)
// Per batch b:  D(512x784) = Wn(512x512) @ Xn(512x784), all row-major.
// fp32 -> split-bf16 (hi+lo) -> v_wmma_f32_16x16x32_bf16 (hi*hi + hi*lo + lo*hi)
// for ~fp32 accuracy at bf16 WMMA rate. Memory-bound: ~207 MB @ 23.3 TB/s ~ 9 us.
// Double-buffered LDS (1 barrier / K-step), async global->LDS for the W tiles.
// ---------------------------------------------------------------------------

typedef __attribute__((ext_vector_type(16))) __bf16 v16bf;
typedef __attribute__((ext_vector_type(8)))  float  v8f;
typedef __attribute__((ext_vector_type(4)))  float  v4f;
typedef __attribute__((ext_vector_type(4)))  int    v4i;

#define GLOBAL_AS __attribute__((address_space(1)))
#define LDS_AS    __attribute__((address_space(3)))

union Frag {
    uint4 q[2];
    v16bf v;
};

#define CCH   512   // channels (K)
#define PN    512   // prototypes (GEMM M)
#define BATCH 64
#define HW    784   // 28*28 (GEMM N)

#define BM 128      // workgroup tile over prototypes
#define BN 128      // workgroup tile over spatial
#define BK 32       // K slab (one bf16 WMMA depth)
#define LDT 40      // LDS row pitch in bf16 elements (80 B, 16B-aligned chunks)

#if __has_builtin(__builtin_amdgcn_global_load_async_to_lds_b128)
#define HAVE_ASYNC 1
#else
#define HAVE_ASYNC 0
#endif

__device__ __forceinline__ void wait_async_all() {
#if HAVE_ASYNC
#if __has_builtin(__builtin_amdgcn_s_wait_asynccnt)
    __builtin_amdgcn_s_wait_asynccnt(0);
#else
    asm volatile("s_wait_asynccnt 0x0" ::: "memory");
#endif
#endif
}

__device__ __forceinline__ void split_bf16(float v, uint16_t& h, uint16_t& l) {
    union { float f; uint32_t u; } c; c.f = v;
    h = (uint16_t)(c.u >> 16);                       // truncate to bf16 (exact residual)
    union { uint32_t u; float f; } hf; hf.u = c.u & 0xffff0000u;
    union { float f; uint32_t u; } cr; cr.f = v - hf.f;
    l = (uint16_t)(cr.u >> 16);
}

// ---- Pass 1: normalize prototype rows, split to bf16 hi/lo ----------------
__global__ __launch_bounds__(256) void proto_norm_split(
    const float* __restrict__ w, uint16_t* __restrict__ whi, uint16_t* __restrict__ wlo)
{
    __shared__ float red[256];
    const int p = blockIdx.x;        // 512 blocks, one prototype row each
    const int t = threadIdx.x;       // 256 threads
    const float* row = w + (size_t)p * CCH;
    float a0 = row[t], a1 = row[t + 256];
    red[t] = a0 * a0 + a1 * a1;
    __syncthreads();
    for (int off = 128; off > 0; off >>= 1) {
        if (t < off) red[t] += red[t + off];
        __syncthreads();
    }
    const float inv = 1.0f / fmaxf(sqrtf(red[0]), 1e-12f);
    uint16_t h, l;
    split_bf16(a0 * inv, h, l);
    whi[(size_t)p * CCH + t] = h;        wlo[(size_t)p * CCH + t] = l;
    split_bf16(a1 * inv, h, l);
    whi[(size_t)p * CCH + t + 256] = h;  wlo[(size_t)p * CCH + t + 256] = l;
}

// ---- Pass 2: inverse L2 norm per (b, hw) column of x ----------------------
__global__ __launch_bounds__(256) void x_colnorm(
    const float* __restrict__ x, float* __restrict__ invn)
{
    const int col = blockIdx.x * 256 + threadIdx.x;      // 0 .. 50175
    if (col >= BATCH * HW) return;
    const int b = col / HW, m = col - b * HW;
    const float* base = x + (size_t)b * CCH * HW + m;
    float s = 0.f;
#pragma unroll 8
    for (int c = 0; c < CCH; ++c) {
        float v = base[(size_t)c * HW];
        s += v * v;
    }
    invn[col] = 1.0f / fmaxf(sqrtf(s), 1e-12f);
}

// ---- Pass 3: batched GEMM via bf16 WMMA, double-buffered LDS --------------
__global__ __launch_bounds__(256) void proto_gemm(
    const float*    __restrict__ x,
    const uint16_t* __restrict__ whi,
    const uint16_t* __restrict__ wlo,
    const float*    __restrict__ invn,
    float*          __restrict__ out)
{
    __shared__ __align__(16) uint16_t sWhi[2][BM * LDT];
    __shared__ __align__(16) uint16_t sWlo[2][BM * LDT];
    __shared__ __align__(16) uint16_t sXhi[2][BN * LDT];
    __shared__ __align__(16) uint16_t sXlo[2][BN * LDT];

    const int tid  = threadIdx.x;
    const int b    = blockIdx.z;
    const int n0   = blockIdx.y * BM;        // prototype tile base
    const int m0   = blockIdx.x * BN;        // spatial tile base

    const int wid   = tid >> 5;              // 8 waves: 2 (n) x 4 (m)
    const int lane  = tid & 31;
    const int wn    = (wid >> 2) * 64;       // wave n offset within tile
    const int wm    = (wid & 3) * 32;        // wave m offset within tile
    const int l16   = lane & 15;
    const int khalf = lane >> 4;

    v8f acc[4][2];
#pragma unroll
    for (int t = 0; t < 4; ++t)
#pragma unroll
        for (int s = 0; s < 2; ++s) acc[t][s] = (v8f){0.f,0.f,0.f,0.f,0.f,0.f,0.f,0.f};

    // ---- W staging map: 2 threads per row, 16 bf16 (32 B) each ----
    const int wrow = tid >> 1;
    const int wh   = (tid & 1) * 16;
    const uint16_t* gwh = whi + (size_t)(n0 + wrow) * CCH + wh;
    const uint16_t* gwl = wlo + (size_t)(n0 + wrow) * CCH + wh;

    // ---- X staging map: float4 along m, K in pairs {r, r+1, r+16, r+17} ----
    const int  mq   = (tid & 31) * 4;        // 0..124
    const int  kb   = (tid >> 5) * 2;        // 0,2,..,14
    const int  mg   = m0 + mq;
    const bool mval = (mg < HW);             // whole float4 valid (784 % 4 == 0)
    v4f inv4 = {0.f, 0.f, 0.f, 0.f};
    if (mval) inv4 = *(const v4f*)&invn[b * HW + mg];
    const float* gxb = x + (size_t)b * CCH * HW + mg;

    v4f xv[2][2];                            // [i: k-group][row r / r+1]

    auto stageW = [&](int k0, int buf) {
#if HAVE_ASYNC
        __builtin_amdgcn_global_load_async_to_lds_b128(
            (GLOBAL_AS v4i*)(gwh + k0),
            (LDS_AS v4i*)&sWhi[buf][wrow * LDT + wh], 0, 0);
        __builtin_amdgcn_global_load_async_to_lds_b128(
            (GLOBAL_AS v4i*)(gwh + k0),
            (LDS_AS v4i*)&sWhi[buf][wrow * LDT + wh], 16, 0);
        __builtin_amdgcn_global_load_async_to_lds_b128(
            (GLOBAL_AS v4i*)(gwl + k0),
            (LDS_AS v4i*)&sWlo[buf][wrow * LDT + wh], 0, 0);
        __builtin_amdgcn_global_load_async_to_lds_b128(
            (GLOBAL_AS v4i*)(gwl + k0),
            (LDS_AS v4i*)&sWlo[buf][wrow * LDT + wh], 16, 0);
#else
        const uint4* ph = (const uint4*)(gwh + k0);
        const uint4* pl = (const uint4*)(gwl + k0);
        uint4 h0 = ph[0], h1 = ph[1];
        uint4 l0 = pl[0], l1 = pl[1];
        *(uint4*)&sWhi[buf][wrow * LDT + wh]     = h0;
        *(uint4*)&sWhi[buf][wrow * LDT + wh + 8] = h1;
        *(uint4*)&sWlo[buf][wrow * LDT + wh]     = l0;
        *(uint4*)&sWlo[buf][wrow * LDT + wh + 8] = l1;
#endif
    };

    auto loadX = [&](int k0) {
#pragma unroll
        for (int i = 0; i < 2; ++i) {
            const int r = k0 + kb + 16 * i;
            if (mval) {
                xv[i][0] = *(const v4f*)(gxb + (size_t)r * HW);
                xv[i][1] = *(const v4f*)(gxb + (size_t)(r + 1) * HW);
            } else {
                xv[i][0] = (v4f){0.f, 0.f, 0.f, 0.f};
                xv[i][1] = (v4f){0.f, 0.f, 0.f, 0.f};
            }
        }
    };

    auto storeX = [&](int buf) {
#pragma unroll
        for (int i = 0; i < 2; ++i) {
            const int r = kb + 16 * i;
            v4f a0 = xv[i][0] * inv4;
            v4f a1 = xv[i][1] * inv4;
#pragma unroll
            for (int j = 0; j < 4; ++j) {
                uint16_t h0, l0p, h1, l1p;
                split_bf16(a0[j], h0, l0p);
                split_bf16(a1[j], h1, l1p);
                *(uint32_t*)&sXhi[buf][(mq + j) * LDT + r] =
                    (uint32_t)h0 | ((uint32_t)h1 << 16);
                *(uint32_t*)&sXlo[buf][(mq + j) * LDT + r] =
                    (uint32_t)l0p | ((uint32_t)l1p << 16);
            }
        }
    };

    // ---- prologue: stage tile 0 into buffer 0 ----
    loadX(0);
    stageW(0, 0);
    storeX(0);

    int cur = 0;
    for (int k0 = 0; k0 < CCH; k0 += BK) {
        wait_async_all();                    // W tile (cur) landed in LDS
        __syncthreads();                     // everyone done staging cur / reading nxt
        const int  nxt  = cur ^ 1;
        const bool more = (k0 + BK) < CCH;
        if (more) {                          // issue next tile's loads under compute
            stageW(k0 + BK, nxt);
            loadX(k0 + BK);
        }

        // -- B fragments (X): lane=col, chunks at khalf*8 and khalf*8+16 --
        Frag bh[2], bl[2];
#pragma unroll
        for (int s = 0; s < 2; ++s) {
            const uint16_t* p  = &sXhi[cur][(wm + s * 16 + l16) * LDT + khalf * 8];
            bh[s].q[0] = *(const uint4*)p;
            bh[s].q[1] = *(const uint4*)(p + 16);
            const uint16_t* p2 = &sXlo[cur][(wm + s * 16 + l16) * LDT + khalf * 8];
            bl[s].q[0] = *(const uint4*)p2;
            bl[s].q[1] = *(const uint4*)(p2 + 16);
        }
        // -- A fragments (W): lane=row, chunks at khalf*8 and khalf*8+16 --
#pragma unroll
        for (int t = 0; t < 4; ++t) {
            Frag ah, al;
            const uint16_t* p  = &sWhi[cur][(wn + t * 16 + l16) * LDT + khalf * 8];
            ah.q[0] = *(const uint4*)p;
            ah.q[1] = *(const uint4*)(p + 16);
            const uint16_t* p2 = &sWlo[cur][(wn + t * 16 + l16) * LDT + khalf * 8];
            al.q[0] = *(const uint4*)p2;
            al.q[1] = *(const uint4*)(p2 + 16);
#pragma unroll
            for (int s = 0; s < 2; ++s) {
                acc[t][s] = __builtin_amdgcn_wmma_f32_16x16x32_bf16(
                    false, ah.v, false, bh[s].v, (short)0, acc[t][s], false, false);
                acc[t][s] = __builtin_amdgcn_wmma_f32_16x16x32_bf16(
                    false, ah.v, false, bl[s].v, (short)0, acc[t][s], false, false);
                acc[t][s] = __builtin_amdgcn_wmma_f32_16x16x32_bf16(
                    false, al.v, false, bh[s].v, (short)0, acc[t][s], false, false);
            }
        }

        if (more) storeX(nxt);               // VALU split + LDS stores after WMMAs
        cur = nxt;
    }

    // -- store D: VGPR j -> row j + 8*khalf, col = lane&15 --
#pragma unroll
    for (int t = 0; t < 4; ++t) {
        const int prow = n0 + wn + t * 16 + khalf * 8;
#pragma unroll
        for (int s = 0; s < 2; ++s) {
            const int mc = m0 + wm + s * 16 + l16;
            if (mc < HW) {
                float* po = out + ((size_t)(b * PN + prow)) * HW + mc;
#pragma unroll
                for (int j = 0; j < 8; ++j) po[(size_t)j * HW] = acc[t][s][j];
            }
        }
    }
}

// ---------------------------------------------------------------------------
extern "C" void kernel_launch(void* const* d_in, const int* in_sizes, int n_in,
                              void* d_out, int out_size, void* d_ws, size_t ws_size,
                              hipStream_t stream) {
    const float* x = (const float*)d_in[0];                 // (64, 512, 28, 28)
    const float* w = (const float*)d_in[1];                 // (512, 512, 1, 1)
    float* out = (float*)d_out;                             // (64, 512, 28, 28)

    char* ws = (char*)d_ws;
    uint16_t* whi = (uint16_t*)(ws);                        // 512 KB
    uint16_t* wlo = (uint16_t*)(ws + (size_t)PN * CCH * 2); // 512 KB
    float*   invn = (float*)(ws + (size_t)PN * CCH * 4);    // 200 KB

    proto_norm_split<<<PN, 256, 0, stream>>>(w, whi, wlo);
    x_colnorm<<<(BATCH * HW + 255) / 256, 256, 0, stream>>>(x, invn);

    dim3 grid((HW + BN - 1) / BN, PN / BM, BATCH);          // (7, 4, 64)
    proto_gemm<<<grid, 256, 0, stream>>>(x, whi, wlo, invn, out);
}
